// CausalSelfAttention_88098369175766
// MI455X (gfx1250) — matrix operand order repaired and checked by reading, not compiled, not run on hardware
//
#include <hip/hip_runtime.h>
#include <cstdint>

// ---------------- problem constants ----------------
#define BATCH 2
#define T_SEQ 2048
#define CDIM  1024
#define HNUM  16
#define DHEAD 64
#define BT    (BATCH * T_SEQ)     // 4096
#define C3    (3 * CDIM)          // 3072

// ---------------- vector types ----------------
typedef __bf16   v16bf __attribute__((ext_vector_type(16)));
typedef __bf16   v8bf  __attribute__((ext_vector_type(8)));
typedef float    v8f   __attribute__((ext_vector_type(8)));
typedef uint32_t u32x4 __attribute__((ext_vector_type(4)));
typedef uint32_t u32x8 __attribute__((ext_vector_type(8)));

union Frag16 { v16bf v; v8bf h[2]; };

// ---------------- CDNA5 helpers ----------------
__device__ __forceinline__ uint32_t lds_byte_off(const void* p) {
  // generic pointers to LDS carry the workgroup-relative byte offset in the low 32 bits
  return (uint32_t)(uintptr_t)p;
}

// Per-lane async 16B copy global->LDS (tracked by ASYNCcnt)
__device__ __forceinline__ void async_copy_b128(uint32_t lds_off, const void* gptr) {
  uint64_t ga = (uint64_t)(uintptr_t)gptr;
  asm volatile("global_load_async_to_lds_b128 %0, %1, off" :: "v"(lds_off), "v"(ga) : "memory");
}
__device__ __forceinline__ void wait_asynccnt0() {
  asm volatile("s_wait_asynccnt 0x0" ::: "memory");
}
__device__ __forceinline__ void wait_dscnt0() {
  asm volatile("s_wait_dscnt 0x0" ::: "memory");
}

// TDM: 2D tile (tile_h rows x tile_w elems of 2 bytes) from row-major tensor
// (row stride = row_stride elems) into contiguous LDS at lds_off.
__device__ __forceinline__ void tdm_load_tile_2d_bf16(uint32_t lds_off, const void* gptr,
                                                      uint32_t tile_w, uint32_t tile_h,
                                                      uint32_t tensor_w, uint32_t tensor_h,
                                                      uint32_t row_stride) {
  uint64_t ga = (uint64_t)(uintptr_t)gptr;
  u32x4 g0;
  g0[0] = 1u;                                                // count=1, user descriptor
  g0[1] = lds_off;                                           // lds_addr (bytes)
  g0[2] = (uint32_t)(ga & 0xFFFFFFFFu);                      // global_addr[31:0]
  g0[3] = (uint32_t)((ga >> 32) & 0x01FFFFFFu) | (2u << 30); // addr[56:32] | type=2
  u32x8 g1;
  g1[0] = (1u << 16);                                        // data_size=1 (2B), wg_mask=0
  g1[1] = (tensor_w & 0xFFFFu) << 16;                        // tensor_dim0 lo16
  g1[2] = (tensor_w >> 16) | ((tensor_h & 0xFFFFu) << 16);   // dim0 hi16 | dim1 lo16
  g1[3] = (tensor_h >> 16) | ((tile_w & 0xFFFFu) << 16);     // dim1 hi16 | tile_dim0
  g1[4] = (tile_h & 0xFFFFu);                                // tile_dim1, tile_dim2=0
  g1[5] = row_stride;                                        // tensor_dim0_stride[31:0]
  g1[6] = 0u;
  g1[7] = 0u;
  asm volatile("tensor_load_to_lds %0, %1" :: "s"(g0), "s"(g1) : "memory");
}

// ---------------- WMMA fragment loaders (ISA 7.12.2), fully vectorized ----------------
// A fragment 16x32: lane row M=(lane&15); K = (e/8)*16 + (lane>=16)*8 + e%8.
__device__ __forceinline__ v16bf load_a_frag(const __bf16* rp, int lane) {
  Frag16 f;
  const int kb = ((lane >> 4) & 1) * 8;
  f.h[0] = *(const v8bf*)(rp + kb);
  f.h[1] = *(const v8bf*)(rp + 16 + kb);
  return f.v;
}
// B fragment 32x16 from K-contiguous ("transposed") tile [ncols][ldk]:
// lane col N=(lane&15); K = (e/8)*16 + (lane>=16)*8 + e%8.
__device__ __forceinline__ v16bf load_b_fragT(const __bf16* tile, int ldk, int col0, int lane) {
  Frag16 f;
  const int n  = lane & 15;
  const int kb = ((lane >> 4) & 1) * 8;
  const __bf16* p = tile + (size_t)(col0 + n) * ldk + kb;
  f.h[0] = *(const v8bf*)(p);
  f.h[1] = *(const v8bf*)(p + 16);
  return f.v;
}

// 16-lane (half-wave) row reductions: C/D rows live within a half-wave
__device__ __forceinline__ float rowmax16(float x) {
  #pragma unroll
  for (int m = 8; m >= 1; m >>= 1) x = fmaxf(x, __shfl_xor(x, m, 32));
  return x;
}
__device__ __forceinline__ float rowsum16(float x) {
  #pragma unroll
  for (int m = 8; m >= 1; m >>= 1) x += __shfl_xor(x, m, 32);
  return x;
}

// ---------------- conversion kernels ----------------
__global__ void cvt_f32_bf16(const float* __restrict__ src, __bf16* __restrict__ dst, int n) {
  int i = blockIdx.x * blockDim.x + threadIdx.x;
  int stride = gridDim.x * blockDim.x;
  for (; i < n; i += stride) dst[i] = (__bf16)src[i];
}
// src [R][Cc] f32 -> dst [Cc][R] bf16 (weights pre-transposed: K-contiguous B tiles)
__global__ void cvt_transpose_f32_bf16(const float* __restrict__ src, __bf16* __restrict__ dst,
                                       int R, int Cc) {
  int i = blockIdx.x * blockDim.x + threadIdx.x;
  int n = R * Cc, stride = gridDim.x * blockDim.x;
  for (; i < n; i += stride) {
    int r = i / Cc, c = i % Cc;                 // coalesced read
    dst[(size_t)c * R + r] = (__bf16)src[i];
  }
}
// V^T: vt[((b*H+h)*D + d)*T + t] = qkv[(b*T+t)*3C + 2C + h*D + d]
__global__ void transpose_v(const __bf16* __restrict__ qkv, __bf16* __restrict__ vt) {
  int i = blockIdx.x * blockDim.x + threadIdx.x;
  int n = BT * CDIM, stride = gridDim.x * blockDim.x;
  for (; i < n; i += stride) {
    int ta = i / CDIM;                          // b*T + t
    int c  = i % CDIM;                          // h*D + d
    int b = ta / T_SEQ, t = ta % T_SEQ;
    int h = c / DHEAD,  d = c % DHEAD;
    vt[(((size_t)b * HNUM + h) * DHEAD + d) * T_SEQ + t] =
        qkv[(size_t)ta * C3 + 2 * CDIM + c];
  }
}

// ---------------- bf16 WMMA GEMM (double-buffered):  C = A * Bt + bias ----------------
// A: [M][K] row-major. Bt: [N][K]. Block tile 128x64, K step 32.
#define BM 128
#define BN 64
#define BK 32

template <bool OUT_F32>
__global__ __launch_bounds__(256)
void gemm_bf16_wmma(const __bf16* __restrict__ A, const __bf16* __restrict__ Bt,
                    const float* __restrict__ bias,
                    float* __restrict__ outF, __bf16* __restrict__ outB,
                    int M, int N, int K) {
  __shared__ __align__(16) __bf16 Atile[2][BM * BK];   // 2 x 8 KB, [m][k]
  __shared__ __align__(16) __bf16 Btile[2][BN * BK];   // 2 x 4 KB, [n][k]

  const int tid  = threadIdx.x;
  const int lane = tid & 31;
  const int wave = __builtin_amdgcn_readfirstlane(tid >> 5);   // scalar: EXEC-safe guards
  const int wm   = wave >> 2;          // 0..1
  const int wn   = wave & 3;           // 0..3
  const int m0   = blockIdx.y * BM;
  const int n0   = blockIdx.x * BN;

  v8f acc[4];
  #pragma unroll
  for (int i = 0; i < 4; ++i)
    #pragma unroll
    for (int j = 0; j < 8; ++j) acc[i][j] = 0.0f;

  // issue one K-tile worth of loads into buffer `bufi`
  auto issue_tile = [&](int k0, int bufi) {
    #pragma unroll
    for (int l = 0; l < 2; ++l) {                 // A: 512 16B chunks, 2/thread (ASYNCcnt)
      int c   = tid + l * 256;
      int row = c >> 2, cg = c & 3;
      async_copy_b128(lds_byte_off(&Atile[bufi][row * BK + cg * 8]),
                      A + (size_t)(m0 + row) * K + k0 + cg * 8);
    }
    if (wave == 0) {                              // B: [64 n][32 k] via TDM (TENSORcnt)
      tdm_load_tile_2d_bf16(lds_byte_off(&Btile[bufi][0]), Bt + (size_t)n0 * K + k0,
                            BK, BN, (uint32_t)K, (uint32_t)N, (uint32_t)K);
    }
  };

  issue_tile(0, 0);                               // prologue
  int buf = 0;
  for (int k0 = 0; k0 < K; k0 += BK, buf ^= 1) {
    wait_asynccnt0();                             // drain this buffer's async copies
    if (wave == 0) __builtin_amdgcn_s_wait_tensorcnt(0);
    __syncthreads();                              // tile visible to all waves

    if (k0 + BK < K) issue_tile(k0 + BK, buf ^ 1);  // overlap next tile with compute

    v16bf bfrag = load_b_fragT(Btile[buf], BK, wn * 16, lane);
    #pragma unroll
    for (int ms = 0; ms < 4; ++ms) {
      const __bf16* arow = &Atile[buf][(size_t)(wm * 64 + ms * 16 + (lane & 15)) * BK];
      v16bf afrag = load_a_frag(arow, lane);
      acc[ms] = __builtin_amdgcn_wmma_f32_16x16x32_bf16(
          false, afrag, false, bfrag, (short)0, acc[ms], false, false);
    }
    __syncthreads();                              // protect buffer against next-next writes
  }

  // epilogue: bias + store (C/D layout: row = v + 8*(lane>=16), col = lane&15)
  #pragma unroll
  for (int ms = 0; ms < 4; ++ms)
    #pragma unroll
    for (int v = 0; v < 8; ++v) {
      int r = m0 + wm * 64 + ms * 16 + v + ((lane >> 4) & 1) * 8;
      int c = n0 + wn * 16 + (lane & 15);
      float val = acc[ms][v] + bias[c];
      if (OUT_F32) outF[(size_t)r * N + c] = val;
      else         outB[(size_t)r * N + c] = (__bf16)val;
    }
}

// ---------------- fused flash attention with T5 rel-bias ----------------
__device__ __forceinline__ float rel_bias_val(int rp, const float* __restrict__ bt, int h) {
  int bucket;
  if (rp < 16) bucket = rp;
  else {
    float t = __logf((float)rp * 0.0625f) * (16.0f / 2.0794415416798357f); // /log(8)*16
    bucket = 16 + (int)t;
    if (bucket > 31) bucket = 31;
  }
  return bt[bucket * HNUM + h];
}

__global__ __launch_bounds__(256)
void attn_flash(const __bf16* __restrict__ qkv, const __bf16* __restrict__ vt,
                const float* __restrict__ bt, __bf16* __restrict__ y) {
  __shared__ __align__(16) __bf16 Ktile[2][32 * 64];   // [key][d], d-contiguous
  __shared__ __align__(16) __bf16 Vtile[2][64 * 32];   // [d][key], key-contiguous
  __shared__ __align__(16) __bf16 Plds[8][16 * 32];    // per-wave P re-layout

  const int tid  = threadIdx.x;
  const int lane = tid & 31;
  const int wave = __builtin_amdgcn_readfirstlane(tid >> 5);   // scalar wave id
  const int b    = blockIdx.z;
  const int h    = blockIdx.y;
  const int q0   = blockIdx.x * 128;
  const int qw   = q0 + wave * 16;       // scalar: this wave's 16 query rows

  __builtin_prefetch(bt, 0, 3);          // bias table -> global_prefetch

  // Q fragments (A layout), D split into two K=32 pieces — vectorized loads
  v16bf aq[2];
  {
    int trow = qw + (lane & 15);
    const __bf16* qrow = qkv + ((size_t)b * T_SEQ + trow) * C3 + h * DHEAD;
    aq[0] = load_a_frag(qrow, lane);
    aq[1] = load_a_frag(qrow + 32, lane);
  }

  float mrow[8], lrow[8];
  v8f acc[4];
  #pragma unroll
  for (int v = 0; v < 8; ++v) { mrow[v] = -1e30f; lrow[v] = 0.0f; }
  #pragma unroll
  for (int f = 0; f < 4; ++f)
    #pragma unroll
    for (int v = 0; v < 8; ++v) acc[f][v] = 0.0f;

  const __bf16* vhead = vt + ((size_t)b * HNUM + h) * DHEAD * T_SEQ;

  auto stage_kv = [&](int key0, int bufi) {
    int krow = tid >> 3, kcg = tid & 7;           // K: 32 rows x 8 chunks
    async_copy_b128(lds_byte_off(&Ktile[bufi][krow * 64 + kcg * 8]),
                    qkv + ((size_t)b * T_SEQ + key0 + krow) * C3 + CDIM + h * DHEAD + kcg * 8);
    int vrow = tid >> 2, vcg = tid & 3;           // V^T: 64 rows x 4 chunks
    async_copy_b128(lds_byte_off(&Vtile[bufi][vrow * 32 + vcg * 8]),
                    vhead + (size_t)vrow * T_SEQ + key0 + vcg * 8);
  };

  const int nkb = (q0 + 128) / 32;       // causal key-block count for this block
  stage_kv(0, 0);                        // prologue
  int buf = 0;
  for (int kb32 = 0; kb32 < nkb; ++kb32, buf ^= 1) {
    const int key0 = kb32 * 32;
    wait_asynccnt0();
    __syncthreads();                     // K/V tiles visible

    if (kb32 + 1 < nkb) stage_kv(key0 + 32, buf ^ 1);   // overlap next K/V with compute

    if (key0 <= qw + 15) {               // scalar branch: EXEC stays all-ones for WMMA
      // ---- S = (Q*0.125)K^T + bias, two 16x16 tiles over 32 keys ----
      v8f s[2];
      #pragma unroll
      for (int sub = 0; sub < 2; ++sub) {
        #pragma unroll
        for (int v = 0; v < 8; ++v) s[sub][v] = 0.0f;
        #pragma unroll
        for (int f = 0; f < 2; ++f) {
          // B = K^T slice: element(k=d, n=key) = Ktile[sub*16+n][f*32+k], d-contiguous
          v16bf bk = load_b_fragT(&Ktile[buf][f * 32], 64, sub * 16, lane);
          s[sub] = __builtin_amdgcn_wmma_f32_16x16x32_bf16(
              false, aq[f], false, bk, (short)0, s[sub], false, false);
        }
        #pragma unroll
        for (int v = 0; v < 8; ++v) {
          int r   = v + ((lane >> 4) & 1) * 8;
          int tq  = qw + r;
          int key = key0 + sub * 16 + (lane & 15);
          if (key > tq) s[sub][v] = -1e30f;
          else s[sub][v] = s[sub][v] * 0.125f + rel_bias_val(tq - key, bt, h);
        }
      }
      // ---- online softmax ----
      #pragma unroll
      for (int v = 0; v < 8; ++v) {
        float tmax  = rowmax16(fmaxf(s[0][v], s[1][v]));
        float mnew  = fmaxf(mrow[v], tmax);
        float alpha = __expf(mrow[v] - mnew);
        float p0 = (s[0][v] <= -5e29f) ? 0.0f : __expf(s[0][v] - mnew);
        float p1 = (s[1][v] <= -5e29f) ? 0.0f : __expf(s[1][v] - mnew);
        lrow[v] = lrow[v] * alpha + rowsum16(p0 + p1);
        mrow[v] = mnew;
        #pragma unroll
        for (int f = 0; f < 4; ++f) acc[f][v] *= alpha;
        // stash P (C/D layout) into per-wave LDS for A-fragment re-layout
        int r = v + ((lane >> 4) & 1) * 8;
        __bf16* pr = &Plds[wave][r * 32 + (lane & 15)];
        pr[0]  = (__bf16)p0;
        pr[16] = (__bf16)p1;
      }
      wait_dscnt0();  // same-wave LDS ops are in-order; fence compiler + DScnt

      // ---- O += P @ V ----
      v16bf ap = load_a_frag(&Plds[wave][(size_t)(lane & 15) * 32], lane);
      #pragma unroll
      for (int f = 0; f < 4; ++f) {
        // B = V slice: element(k=key, n=d) = Vtile[f*16+n][k], key-contiguous
        v16bf bv = load_b_fragT(Vtile[buf], 32, f * 16, lane);
        acc[f] = __builtin_amdgcn_wmma_f32_16x16x32_bf16(
            false, ap, false, bv, (short)0, acc[f], false, false);
      }
    }
    __syncthreads();                     // protect buffer before next-next staging
  }

  // epilogue: O / l -> y[b, t, h*64 + d]
  #pragma unroll
  for (int f = 0; f < 4; ++f)
    #pragma unroll
    for (int v = 0; v < 8; ++v) {
      int r = v + ((lane >> 4) & 1) * 8;
      int t = qw + r;
      int d = f * 16 + (lane & 15);
      float val = acc[f][v] / lrow[v];
      y[((size_t)b * T_SEQ + t) * CDIM + h * DHEAD + d] = (__bf16)val;
    }
}

// ---------------- host launcher ----------------
extern "C" void kernel_launch(void* const* d_in, const int* in_sizes, int n_in,
                              void* d_out, int out_size, void* d_ws, size_t ws_size,
                              hipStream_t stream) {
  (void)in_sizes; (void)n_in; (void)out_size; (void)ws_size;
  const float* x          = (const float*)d_in[0];
  const float* W_attn     = (const float*)d_in[1];
  const float* b_attn     = (const float*)d_in[2];
  const float* W_proj     = (const float*)d_in[3];
  const float* b_proj     = (const float*)d_in[4];
  const float* bias_table = (const float*)d_in[5];

  // workspace layout (bf16, 56 MB total)
  __bf16* xb   = (__bf16*)d_ws;
  __bf16* WabT = xb   + (size_t)BT * CDIM;       // W_attn^T [3C][C]
  __bf16* WpbT = WabT + (size_t)C3 * CDIM;       // W_proj^T [C][C]
  __bf16* qkvb = WpbT + (size_t)CDIM * CDIM;     // qkv bf16 [BT][3C]
  __bf16* yb   = qkvb + (size_t)BT * C3;         // attention out [BT][C]
  __bf16* vtb  = yb   + (size_t)BT * CDIM;       // V^T [B][H][D][T]

  cvt_f32_bf16<<<1024, 256, 0, stream>>>(x, xb, BT * CDIM);
  cvt_transpose_f32_bf16<<<1024, 256, 0, stream>>>(W_attn, WabT, CDIM, C3);
  cvt_transpose_f32_bf16<<<512,  256, 0, stream>>>(W_proj, WpbT, CDIM, CDIM);

  // QKV projection: [4096,1024] x [1024,3072] -> bf16 qkv
  dim3 g1(C3 / BN, BT / BM);   // 48 x 32
  gemm_bf16_wmma<false><<<g1, 256, 0, stream>>>(xb, WabT, b_attn,
                                                nullptr, qkvb, BT, C3, CDIM);

  // V^T for key-contiguous V fragments
  transpose_v<<<1024, 256, 0, stream>>>(qkvb, vtb);

  // fused causal attention with T5 rel bias
  dim3 ga(T_SEQ / 128, HNUM, BATCH);  // 16 x 16 x 2
  attn_flash<<<ga, 256, 0, stream>>>(qkvb, vtb, bias_table, yb);

  // output projection: [4096,1024] x [1024,1024] -> f32 out
  dim3 g2(CDIM / BN, BT / BM); // 16 x 32
  gemm_bf16_wmma<true><<<g2, 256, 0, stream>>>(yb, WpbT, b_proj,
                                               (float*)d_out, nullptr, BT, CDIM, CDIM);
}
